// DeformableConv2d_29257317220484
// MI455X (gfx1250) — matrix operand order repaired and checked by reading, not compiled, not run on hardware
//
#include <hip/hip_runtime.h>
#include <hip/hip_bf16.h>
#include <hip/hip_fp16.h>

// ---------------------------------------------------------------------------
// DCNv2 (modulated deformable conv) for MI455X / gfx1250, wave32 + WMMA.
//   B=8, Cin=Cout=64, H=W=128, 3x3, stride=1, pad=1, dil=1
// Pipeline:
//   K1: offset(18ch)+mask(9ch) 3x3 conv, fp32 VALU, weights transposed in LDS
//   K2: convert main conv weight [64][576] f32 -> f16 padded [64][584]
//   K3: fused bilinear-gather -> LDS column tile -> WMMA f16 GEMM (M=64,N=32/blk)
// ---------------------------------------------------------------------------

typedef _Float16 v16h __attribute__((ext_vector_type(16)));
typedef float    v8f  __attribute__((ext_vector_type(8)));

#define HGT     128
#define WID     128
#define HW      16384      // 128*128
#define CIN     64
#define COUT    64
#define NB      8
#define K2      9
#define KDIM    576        // CIN*K2
#define COLSTR  584        // KDIM padded to multiple of 8 halves (16B)
#define NPIX    131072     // NB*HW

// ----------------------------- Kernel 1 ------------------------------------
// 27-channel 3x3 conv (18 offset + 9 modulator->sigmoid).
// LDS weights transposed: wT[k = c*9+tap][oc], oc-stride padded to 28 floats.
__global__ __launch_bounds__(256) void conv27_kernel(
    const float* __restrict__ x,
    const float* __restrict__ ow, const float* __restrict__ ob,
    const float* __restrict__ mw, const float* __restrict__ mb,
    float* __restrict__ offs, float* __restrict__ mask)
{
    __shared__ __align__(16) float wT[KDIM * 28];   // 64512 B
    const int tid = threadIdx.x;

    for (int i = tid; i < 18 * KDIM; i += 256) {    // offset weights -> oc 0..17
        int oc = i / KDIM, k = i % KDIM;
        wT[k * 28 + oc] = ow[i];
    }
    for (int i = tid; i < 9 * KDIM; i += 256) {     // mod weights -> oc 18..26
        int oc = i / KDIM, k = i % KDIM;
        wT[k * 28 + 18 + oc] = mw[i];
    }
    __syncthreads();

    const int p  = blockIdx.x * 256 + tid;          // pixel (exactly NPIX threads)
    const int b  = p >> 14;
    const int hw = p & (HW - 1);
    const int h  = hw >> 7;
    const int w  = hw & (WID - 1);

    float acc[27];
#pragma unroll
    for (int oc = 0; oc < 18; ++oc) acc[oc] = ob[oc];
#pragma unroll
    for (int j = 0; j < 9; ++j)     acc[18 + j] = mb[j];

    const float* xb = x + b * CIN * HW;
    for (int tap = 0; tap < 9; ++tap) {
        const int yy = h + tap / 3 - 1;
        const int xx = w + tap % 3 - 1;
        if (yy < 0 || yy >= HGT || xx < 0 || xx >= WID) continue;
        const float* xp = xb + yy * WID + xx;
        for (int c = 0; c < CIN; ++c) {
            const float xv = xp[c * HW];
            const float4* wp = (const float4*)&wT[(c * 9 + tap) * 28];
            float wv[28];
#pragma unroll
            for (int q = 0; q < 7; ++q) ((float4*)wv)[q] = wp[q];
#pragma unroll
            for (int oc = 0; oc < 27; ++oc) acc[oc] += xv * wv[oc];
        }
    }

#pragma unroll
    for (int oc = 0; oc < 18; ++oc)
        offs[(b * 18 + oc) * HW + hw] = acc[oc];
#pragma unroll
    for (int j = 0; j < 9; ++j)
        mask[(b * 9 + j) * HW + hw] = 1.0f / (1.0f + __expf(-acc[18 + j]));
}

// ----------------------------- Kernel 2 ------------------------------------
__global__ __launch_bounds__(256) void cvt_weight_kernel(
    const float* __restrict__ wsrc, _Float16* __restrict__ wdst)
{
    const int i = blockIdx.x * 256 + threadIdx.x;
    if (i < COUT * KDIM) {
        const int m = i / KDIM, k = i % KDIM;
        wdst[m * COLSTR + k] = (_Float16)wsrc[i];
    }
}

// ----------------------------- Kernel 3 ------------------------------------
// Per block: 32 pixels. Gather deformed/modulated samples (f16) into LDS
// column tile col[n][k], k = c*9+tap (matches weight.reshape(Cout,Cin,K2)).
// Then 8 waves x 1 WMMA subtile each compute the 64x32 output tile.
__global__ __launch_bounds__(256) void dcn_wmma_kernel(
    const float* __restrict__ x, const _Float16* __restrict__ wH,
    const float* __restrict__ offs, const float* __restrict__ mask,
    float* __restrict__ out)
{
    __shared__ __align__(16) _Float16 col[32 * COLSTR];   // 37376 B
    const int tid  = threadIdx.x;
    const int base = blockIdx.x * 32;

    // ---- gather phase: 32 pixels x 9 taps over 256 threads -----------------
    for (int it = tid; it < 32 * 9; it += 256) {
        const int n = it / 9, tap = it % 9;
        const int p  = base + n;
        const int b  = p >> 14;
        const int hw = p & (HW - 1);
        const int h  = hw >> 7;
        const int w  = hw & (WID - 1);

        const float oy = offs[(b * 18 + 2 * tap)     * HW + hw];
        const float ox = offs[(b * 18 + 2 * tap + 1) * HW + hw];
        const float mv = mask[(b * 9 + tap) * HW + hw];

        const float yf = (float)(h + tap / 3 - 1) + oy;
        const float xf = (float)(w + tap % 3 - 1) + ox;
        const float y0f = floorf(yf), x0f = floorf(xf);
        const float ly = yf - y0f, lx = xf - x0f;
        const int y0 = (int)y0f, x0 = (int)x0f;
        const int y1 = y0 + 1,   x1 = x0 + 1;

        const bool vy0 = (y0 >= 0) & (y0 < HGT), vy1 = (y1 >= 0) & (y1 < HGT);
        const bool vx0 = (x0 >= 0) & (x0 < WID), vx1 = (x1 >= 0) & (x1 < WID);
        const float w00 = (vy0 & vx0) ? (1.0f - ly) * (1.0f - lx) * mv : 0.0f;
        const float w01 = (vy0 & vx1) ? (1.0f - ly) * lx          * mv : 0.0f;
        const float w10 = (vy1 & vx0) ? ly          * (1.0f - lx) * mv : 0.0f;
        const float w11 = (vy1 & vx1) ? ly          * lx          * mv : 0.0f;

        const int cy0 = min(max(y0, 0), HGT - 1), cy1 = min(max(y1, 0), HGT - 1);
        const int cx0 = min(max(x0, 0), WID - 1), cx1 = min(max(x1, 0), WID - 1);
        const int i00 = cy0 * WID + cx0, i01 = cy0 * WID + cx1;
        const int i10 = cy1 * WID + cx0, i11 = cy1 * WID + cx1;

        const float* xb = x + b * CIN * HW;
        _Float16* cp = &col[n * COLSTR + tap];
        for (int c = 0; c < CIN; ++c) {
            const int o = c * HW;
            const float v = w00 * xb[i00 + o] + w01 * xb[i01 + o]
                          + w10 * xb[i10 + o] + w11 * xb[i11 + o];
            cp[c * 9] = (_Float16)v;
        }
    }
    __syncthreads();

    // ---- GEMM phase: D = W(64x576) x col^T(576x32), f16 WMMA, f32 acc ------
    const int wv   = tid >> 5;          // wave 0..7
    const int l    = tid & 31;          // lane
    const int msub = (wv & 3) * 16;     // Cout subtile
    const int nsub = (wv >> 2) * 16;    // pixel subtile
    const int lr   = l & 15;
    const int lh   = l >> 4;

    // A (16x32 f16): lane holds row lr, chunks at k+(lh?8:0) and +16.
    // B (32x16 f16): lane holds col lr, chunks at k+(lh?8:0) and +16.
    const _Float16* wrow = wH  + (msub + lr) * COLSTR + lh * 8;
    const _Float16* crow = col + (nsub + lr) * COLSTR + lh * 8;

    v8f acc = {0.f, 0.f, 0.f, 0.f, 0.f, 0.f, 0.f, 0.f};
    for (int ks = 0; ks < KDIM; ks += 32) {
        union { v16h v; float4 q[2]; } afrag, bfrag;
        afrag.q[0] = *(const float4*)(wrow + ks);
        afrag.q[1] = *(const float4*)(wrow + ks + 16);
        bfrag.q[0] = *(const float4*)(crow + ks);
        bfrag.q[1] = *(const float4*)(crow + ks + 16);
        acc = __builtin_amdgcn_wmma_f32_16x16x32_f16(
                  false, afrag.v, false, bfrag.v, (short)0, acc, false, false);
    }

    // D layout: acc[v] -> row (msub + v + 8*lh), col (nsub + lr)
    const int p  = base + nsub + lr;
    const int b  = p >> 14;
    const int hw = p & (HW - 1);
    float* ob = out + (b * COUT + msub + lh * 8) * HW + hw;
#pragma unroll
    for (int v = 0; v < 8; ++v) ob[v * HW] = acc[v];
}

// ----------------------------- Launcher ------------------------------------
extern "C" void kernel_launch(void* const* d_in, const int* in_sizes, int n_in,
                              void* d_out, int out_size, void* d_ws, size_t ws_size,
                              hipStream_t stream)
{
    const float* x   = (const float*)d_in[0];   // [8,64,128,128]
    const float* ow  = (const float*)d_in[1];   // [18,64,3,3]
    const float* ob  = (const float*)d_in[2];   // [18]
    const float* mw  = (const float*)d_in[3];   // [9,64,3,3]
    const float* mb  = (const float*)d_in[4];   // [9]
    const float* wgt = (const float*)d_in[5];   // [64,64,3,3]
    float* out = (float*)d_out;                 // [8,64,128,128]

    // workspace layout
    float*     offs = (float*)d_ws;                       // 8*18*16384 f32
    float*     mask = offs + NB * 18 * HW;                // 8*9*16384  f32
    _Float16*  wH   = (_Float16*)(mask + NB * 9 * HW);    // 64*584     f16

    conv27_kernel<<<NPIX / 256, 256, 0, stream>>>(x, ow, ob, mw, mb, offs, mask);
    cvt_weight_kernel<<<(COUT * KDIM + 255) / 256, 256, 0, stream>>>(wgt, wH);
    dcn_wmma_kernel<<<NPIX / 32, 256, 0, stream>>>(x, wH, offs, mask, out);
}
